// GAT_13340168421835
// MI455X (gfx1250) — matrix operand order repaired and checked by reading, not compiled, not run on hardware
//
#include <hip/hip_runtime.h>

// ---------------------------------------------------------------------------
// GAT (2-layer) for MI455X / gfx1250.  N=50000, E=800000, IN=128, H1=4, C1=32,
// C2=16.  GEMMs via v_wmma_f32_16x16x32_f16; edge softmax/aggregation via
// L2-resident f32 atomics (whole working set < 192MB L2).
// ---------------------------------------------------------------------------

#define H1 4
#define C1 32
#define F1 128          // H1*C1 == IN_C
#define C2 16

typedef __attribute__((ext_vector_type(16))) _Float16 v16h;
typedef __attribute__((ext_vector_type(8)))  _Float16 v8h;
typedef __attribute__((ext_vector_type(8)))  float    v8f;

// ---- float <-> order-preserving uint (for atomicMax on floats) ------------
__device__ __forceinline__ unsigned ford(float f) {
    unsigned u = __float_as_uint(f);
    return (u & 0x80000000u) ? ~u : (u | 0x80000000u);
}
__device__ __forceinline__ float fdeord(unsigned u) {
    unsigned b = (u & 0x80000000u) ? (u & 0x7FFFFFFFu) : ~u;
    return __uint_as_float(b);
}

// ---- WMMA fragment loaders ------------------------------------------------
// A-matrix 16x32 f16 (ISA table): lane L -> row M=L&15; K-offset 8*(L>>4);
// elements 0..7 = K[koff..koff+7], elements 8..15 = K[koff+16..koff+23].
__device__ __forceinline__ v16h load_afrag(const _Float16* arow_kb, int lane) {
    const _Float16* p = arow_kb + ((lane >> 4) << 3);
    union { v16h v; v8h h[2]; } u;
    u.h[0] = *reinterpret_cast<const v8h*>(p);
    u.h[1] = *reinterpret_cast<const v8h*>(p + 16);
    return u.v;
}
// B-matrix 32x16 f16: lane L -> column N=L&15; K-offset 8*(L>>4);
// elements 0..7 = K[koff..koff+7], elements 8..15 = K[koff+16..koff+23].
__device__ __forceinline__ v16h load_bfrag(const _Float16* bcol_kb, int lane) {
    const _Float16* p = bcol_kb + ((lane >> 4) << 3);
    union { v16h v; v8h h[2]; } u;
    u.h[0] = *reinterpret_cast<const v8h*>(p);
    u.h[1] = *reinterpret_cast<const v8h*>(p + 16);
    return u.v;
}

// ---- utility kernels ------------------------------------------------------
__global__ void fill_u32_kernel(unsigned* __restrict__ p, unsigned v, long long n) {
    long long i = (long long)blockIdx.x * blockDim.x + threadIdx.x;
    if (i < n) p[i] = v;
}

__global__ void cvt_f16_kernel(const float* __restrict__ in, _Float16* __restrict__ out, int n) {
    int i = blockIdx.x * blockDim.x + threadIdx.x;
    if (i < n) out[i] = (_Float16)in[i];
}

// in: [K, Nc] row-major f32  ->  out: [Nc, K] f16 (column of W becomes a row)
__global__ void transpose_cvt_kernel(const float* __restrict__ in, _Float16* __restrict__ out,
                                     int K, int Nc) {
    int i = blockIdx.x * blockDim.x + threadIdx.x;
    if (i >= K * Nc) return;
    int k = i / Nc, n = i % Nc;
    out[n * K + k] = (_Float16)in[k * Nc + n];
}

// ---- GEMM1: h1[M,128] = xh[M,128] @ W1 (w1t = [128 cols][128 k] f16) -------
// grid = M/16 blocks of 256 threads; wave w (0..7) owns column tile n0=16w.
__global__ void gemm1_kernel(const _Float16* __restrict__ Ah,
                             const _Float16* __restrict__ Bt,
                             float* __restrict__ D) {
    int wave = threadIdx.x >> 5;
    int lane = threadIdx.x & 31;
    int m0 = blockIdx.x << 4;
    int n0 = wave << 4;
    int row = m0 + (lane & 15);
    int col = n0 + (lane & 15);
    const _Float16* arow = Ah + (size_t)row * F1;
    const _Float16* bcol = Bt + (size_t)col * F1;
    v8f acc = {};
#pragma unroll
    for (int kb = 0; kb < 4; ++kb) {
        v16h a = load_afrag(arow + kb * 32, lane);
        v16h b = load_bfrag(bcol + kb * 32, lane);
        acc = __builtin_amdgcn_wmma_f32_16x16x32_f16(false, a, false, b,
                                                     (short)0, acc, false, false);
    }
    int mbase = m0 + ((lane >> 4) << 3);
#pragma unroll
    for (int v = 0; v < 8; ++v)
        D[(size_t)(mbase + v) * F1 + col] = acc[v];
}

// ---- GEMM2: h2[M,16] = hact[M,128] @ W2 (w2t = [16 cols][128 k] f16) -------
__global__ void gemm2_kernel(const _Float16* __restrict__ Ah,
                             const _Float16* __restrict__ Bt,
                             float* __restrict__ D, int Mtiles) {
    int wave = threadIdx.x >> 5;
    int lane = threadIdx.x & 31;
    int tile = blockIdx.x * 8 + wave;
    if (tile >= Mtiles) return;          // wave-uniform branch: EXEC stays all-1
    int m0 = tile << 4;
    int row = m0 + (lane & 15);
    int col = lane & 15;
    const _Float16* arow = Ah + (size_t)row * F1;
    const _Float16* bcol = Bt + (size_t)col * F1;
    v8f acc = {};
#pragma unroll
    for (int kb = 0; kb < 4; ++kb) {
        v16h a = load_afrag(arow + kb * 32, lane);
        v16h b = load_bfrag(bcol + kb * 32, lane);
        acc = __builtin_amdgcn_wmma_f32_16x16x32_f16(false, a, false, b,
                                                     (short)0, acc, false, false);
    }
    int mbase = m0 + ((lane >> 4) << 3);
#pragma unroll
    for (int v = 0; v < 8; ++v)
        D[(size_t)(mbase + v) * C2 + col] = acc[v];
}

// ---- attention logits -----------------------------------------------------
__global__ void logits1_kernel(const float* __restrict__ h1,
                               const float* __restrict__ att_s,
                               const float* __restrict__ att_d,
                               float* __restrict__ as_, float* __restrict__ ad_, int Nn) {
    int i = blockIdx.x * blockDim.x + threadIdx.x;   // (n*H1 + h)
    if (i >= Nn * H1) return;
    int n = i >> 2, h = i & 3;
    const float* hp = h1 + (size_t)n * F1 + h * C1;
    const float* sp = att_s + h * C1;
    const float* dp = att_d + h * C1;
    float s = 0.f, d = 0.f;
#pragma unroll
    for (int c = 0; c < C1; ++c) { float v = hp[c]; s += v * sp[c]; d += v * dp[c]; }
    as_[i] = s; ad_[i] = d;
}

__global__ void logits2_kernel(const float* __restrict__ h2,
                               const float* __restrict__ att_s,
                               const float* __restrict__ att_d,
                               float* __restrict__ as_, float* __restrict__ ad_, int Nn) {
    int n = blockIdx.x * blockDim.x + threadIdx.x;
    if (n >= Nn) return;
    float s = 0.f, d = 0.f;
#pragma unroll
    for (int c = 0; c < C2; ++c) { float v = h2[(size_t)n * C2 + c]; s += v * att_s[c]; d += v * att_d[c]; }
    as_[n] = s; ad_[n] = d;
}

// ---- edge passes ----------------------------------------------------------
template <int HN>
__global__ void edge_max_kernel(const int* __restrict__ ei, int E_, int Nn,
                                const float* __restrict__ as_, const float* __restrict__ ad_,
                                float* __restrict__ alpha, unsigned* __restrict__ amax) {
    int i = blockIdx.x * blockDim.x + threadIdx.x;   // e*HN + h
    int ET = (E_ + Nn) * HN;
    if (i >= ET) return;
    int e = i / HN, h = i - e * HN;
    int s, d;
    if (e < E_) { s = ei[e]; d = ei[E_ + e]; } else { s = d = e - E_; }
    float a = as_[s * HN + h] + ad_[d * HN + h];
    a = (a > 0.f) ? a : 0.2f * a;                    // leaky relu
    alpha[i] = a;
    atomicMax(&amax[d * HN + h], ford(a));
}

template <int HN>
__global__ void edge_exp_kernel(const int* __restrict__ ei, int E_, int Nn,
                                float* __restrict__ alpha, const unsigned* __restrict__ amax,
                                float* __restrict__ den) {
    int i = blockIdx.x * blockDim.x + threadIdx.x;
    int ET = (E_ + Nn) * HN;
    if (i >= ET) return;
    int e = i / HN, h = i - e * HN;
    int d = (e < E_) ? ei[E_ + e] : e - E_;
    float m = fdeord(amax[d * HN + h]);
    float ex = expf(alpha[i] - m);
    alpha[i] = ex;                                   // becomes the numerator
    atomicAdd(&den[d * HN + h], ex);
}

// wave-per-edge; lane owns 4 contiguous channels (head = lane/8)
__global__ void edge_agg1_kernel(const int* __restrict__ ei, int E_, int Nn,
                                 const float* __restrict__ h1, const float* __restrict__ ex,
                                 const float* __restrict__ den, float* __restrict__ out1) {
    int wave = threadIdx.x >> 5, lane = threadIdx.x & 31;
    int e = blockIdx.x * (blockDim.x >> 5) + wave;
    if (e >= E_ + Nn) return;
    int s, d;
    if (e < E_) { s = ei[e]; d = ei[E_ + e]; } else { s = d = e - E_; }
    int h = lane >> 3;
    float coef = ex[e * H1 + h] / den[d * H1 + h];
    float4 v = reinterpret_cast<const float4*>(h1 + (size_t)s * F1)[lane];
    float* op = out1 + (size_t)d * F1 + lane * 4;
    atomicAdd(op + 0, v.x * coef);
    atomicAdd(op + 1, v.y * coef);
    atomicAdd(op + 2, v.z * coef);
    atomicAdd(op + 3, v.w * coef);
}

__global__ void edge_agg2_kernel(const int* __restrict__ ei, int E_, int Nn,
                                 const float* __restrict__ h2, const float* __restrict__ ex,
                                 const float* __restrict__ den, float* __restrict__ out) {
    long long i = (long long)blockIdx.x * blockDim.x + threadIdx.x;
    long long ET = (long long)(E_ + Nn) * C2;
    if (i >= ET) return;
    int e = (int)(i >> 4), c = (int)(i & 15);
    int s, d;
    if (e < E_) { s = ei[e]; d = ei[E_ + e]; } else { s = d = e - E_; }
    float coef = ex[e] / den[d];
    atomicAdd(&out[(size_t)d * C2 + c], h2[(size_t)s * C2 + c] * coef);
}

// ---- epilogues ------------------------------------------------------------
__global__ void bias_elu1_kernel(const float* __restrict__ o1, const float* __restrict__ b1,
                                 _Float16* __restrict__ hact, int Nn) {
    int i = blockIdx.x * blockDim.x + threadIdx.x;
    if (i >= Nn * F1) return;
    float v = o1[i] + b1[i & (F1 - 1)];
    v = (v > 0.f) ? v : (expf(v) - 1.f);             // ELU(alpha=1)
    hact[i] = (_Float16)v;
}

__global__ void bias2_kernel(float* __restrict__ out, const float* __restrict__ b2, int Nn) {
    int i = blockIdx.x * blockDim.x + threadIdx.x;
    if (i < Nn * C2) out[i] += b2[i & (C2 - 1)];
}

// ---------------------------------------------------------------------------
extern "C" void kernel_launch(void* const* d_in, const int* in_sizes, int n_in,
                              void* d_out, int out_size, void* d_ws, size_t ws_size,
                              hipStream_t stream) {
    const float* x    = (const float*)d_in[0];
    const int*   ei   = (const int*)d_in[1];
    const float* W1   = (const float*)d_in[2];
    const float* atS1 = (const float*)d_in[3];
    const float* atD1 = (const float*)d_in[4];
    const float* b1   = (const float*)d_in[5];
    const float* W2   = (const float*)d_in[6];
    const float* atS2 = (const float*)d_in[7];
    const float* atD2 = (const float*)d_in[8];
    const float* b2   = (const float*)d_in[9];
    float* out = (float*)d_out;

    const int Nn = in_sizes[0] / F1;          // 50000 (multiple of 16)
    const int Ee = in_sizes[1] / 2;           // 800000
    const int ET = Ee + Nn;                   // with self loops

    // ---- workspace carve-out (256B aligned) ----
    char* wp = (char*)d_ws;
    auto carve = [&](size_t bytes) { char* r = wp; wp += (bytes + 255) & ~(size_t)255; return r; };
    _Float16* xh  = (_Float16*)carve((size_t)Nn * F1 * 2);
    _Float16* w1t = (_Float16*)carve((size_t)F1 * F1 * 2);
    float*    h1  = (float*)   carve((size_t)Nn * F1 * 4);
    float*    aS1 = (float*)   carve((size_t)Nn * H1 * 4);
    float*    aD1 = (float*)   carve((size_t)Nn * H1 * 4);
    unsigned* mx1 = (unsigned*)carve((size_t)Nn * H1 * 4);
    float*    dn1 = (float*)   carve((size_t)Nn * H1 * 4);
    float*    ex1 = (float*)   carve((size_t)ET * H1 * 4);
    float*    o1  = (float*)   carve((size_t)Nn * F1 * 4);
    _Float16* w2t = (_Float16*)carve((size_t)F1 * C2 * 2);
    float*    h2  = (float*)   carve((size_t)Nn * C2 * 4);
    float*    aS2 = (float*)   carve((size_t)Nn * 4);
    float*    aD2 = (float*)   carve((size_t)Nn * 4);
    unsigned* mx2 = (unsigned*)carve((size_t)Nn * 4);
    float*    dn2 = (float*)   carve((size_t)Nn * 4);
    float*    ex2 = ex1;                      // ex1 dead before layer-2 edge passes
    _Float16* hact = xh;                      // xh dead after GEMM1

    const int B = 256;
    auto cdiv = [](long long a, long long b) { return (int)((a + b - 1) / b); };

    // ===== Layer 1 =====
    cvt_f16_kernel<<<cdiv((long long)Nn * F1, B), B, 0, stream>>>(x, xh, Nn * F1);
    transpose_cvt_kernel<<<cdiv(F1 * F1, B), B, 0, stream>>>(W1, w1t, F1, F1);
    gemm1_kernel<<<Nn / 16, B, 0, stream>>>(xh, w1t, h1);
    logits1_kernel<<<cdiv((long long)Nn * H1, B), B, 0, stream>>>(h1, atS1, atD1, aS1, aD1, Nn);

    fill_u32_kernel<<<cdiv((long long)Nn * H1, B), B, 0, stream>>>(mx1, 0u, (long long)Nn * H1);
    fill_u32_kernel<<<cdiv((long long)Nn * H1, B), B, 0, stream>>>((unsigned*)dn1, 0u, (long long)Nn * H1);
    fill_u32_kernel<<<cdiv((long long)Nn * F1, B), B, 0, stream>>>((unsigned*)o1, 0u, (long long)Nn * F1);

    edge_max_kernel<H1><<<cdiv((long long)ET * H1, B), B, 0, stream>>>(ei, Ee, Nn, aS1, aD1, ex1, mx1);
    edge_exp_kernel<H1><<<cdiv((long long)ET * H1, B), B, 0, stream>>>(ei, Ee, Nn, ex1, mx1, dn1);
    edge_agg1_kernel<<<cdiv(ET, B / 32), B, 0, stream>>>(ei, Ee, Nn, h1, ex1, dn1, o1);

    bias_elu1_kernel<<<cdiv((long long)Nn * F1, B), B, 0, stream>>>(o1, b1, hact, Nn);

    // ===== Layer 2 =====
    transpose_cvt_kernel<<<cdiv(F1 * C2, B), B, 0, stream>>>(W2, w2t, F1, C2);
    gemm2_kernel<<<cdiv(Nn / 16, 8), B, 0, stream>>>(hact, w2t, h2, Nn / 16);
    logits2_kernel<<<cdiv(Nn, B), B, 0, stream>>>(h2, atS2, atD2, aS2, aD2, Nn);

    fill_u32_kernel<<<cdiv(Nn, B), B, 0, stream>>>(mx2, 0u, Nn);
    fill_u32_kernel<<<cdiv(Nn, B), B, 0, stream>>>((unsigned*)dn2, 0u, Nn);
    fill_u32_kernel<<<cdiv((long long)Nn * C2, B), B, 0, stream>>>((unsigned*)out, 0u, (long long)Nn * C2);

    edge_max_kernel<1><<<cdiv(ET, B), B, 0, stream>>>(ei, Ee, Nn, aS2, aD2, ex2, mx2);
    edge_exp_kernel<1><<<cdiv(ET, B), B, 0, stream>>>(ei, Ee, Nn, ex2, mx2, dn2);
    edge_agg2_kernel<<<cdiv((long long)ET * C2, B), B, 0, stream>>>(ei, Ee, Nn, h2, ex2, dn2, out);

    bias2_kernel<<<cdiv((long long)Nn * C2, B), B, 0, stream>>>(out, b2, Nn);
    (void)ws_size; (void)n_in; (void)out_size;
}
